// GraphRelationPriorNet_38087769981452
// MI455X (gfx1250) — matrix operand order repaired and checked by reading, not compiled, not run on hardware
//
#include <hip/hip_runtime.h>
#include <hip/hip_bf16.h>

// ---------------------------------------------------------------------------
// GraphRelationPriorNet for MI455X (gfx1250, wave32, WMMA)
//
// Shapes: B=4, T=256, D=256, H=4, DK=64, NUM_REL=64
//
// Pipeline (all dense-GEMM parts use v_wmma_f32_16x16x32_f16; one wave per
// 32x64 output tile = 2x4 grid of 16x16 WMMA accumulators, 8 WMMA / K-step):
//   1. q,k,v   = X @ W{q,k,v}^T + b             (3x GEMM 1024x256x256)
//   2. qr[b,h,i,r] = <q[b,i,h,:], rel_k[r,h,:]> (batched GEMM 256x64x64)
//   3. qk[b,h,i,j] = <q_i, k_j>                 (batched GEMM 256x256x64)
//   4. scores = (qk + gather(qr, sp)) / 8       (elementwise + gather)
//   5. softmax over j                           (block reduction)
//   6. aw[b,h,i,r] = sum_{j: sp==r} attn        (LDS ds_add_f32 histogram)
//   7. agg = attn @ V  +  aw @ rel_v            (2x batched GEMM, accumulate)
//   8. proj = agg @ Wo^T + bo                   (GEMM 1024x256x256)
//   9. out = relu(LayerNorm(proj + X))          (block reduction)
// ---------------------------------------------------------------------------

typedef __attribute__((ext_vector_type(16))) _Float16 v16h;
typedef __attribute__((ext_vector_type(8)))  float    v8f;

#define B_ 4
#define T_ 256
#define D_ 256
#define H_ 4
#define DK_ 64
#define R_ 64

// ---------------------------------------------------------------------------
// Generic batched WMMA GEMM:  OUT[m,n] (+)= scale * sum_k A[m,k] * Bsel + bias
//   TRANS_B=1 : Bsel = Bm[n*ldb + k]   (OUT = A @ B^T, both row-major in k)
//   TRANS_B=0 : Bsel = Bm[k*ldb + n]   (OUT = A @ B)
// Batch z decomposed as z1=z/z2n, z2=z%z2n with separate strides, so per-(b,h)
// slices (head offset 64, batch offset T*D) are expressible.
// One wave computes a 32x64 tile: MI=2 x NJ=4 accumulators, 8 WMMA per K-step.
// Requires M%32==0, N%64==0, K%32==0 (true for every GEMM in this net).
// ---------------------------------------------------------------------------
#define MI 2
#define NJ 4

template<bool TRANS_B, bool ACCUM, bool HAS_BIAS>
__global__ void __launch_bounds__(32)
wmma_gemm_f16(const float* __restrict__ A, const float* __restrict__ Bm,
              const float* __restrict__ bias, float* __restrict__ OUT,
              int K, int lda, int ldb, int ldo,
              long offA1, long offA2, long offB1, long offB2,
              long offO1, long offO2, int z2n, float scale)
{
    const int lane = threadIdx.x;            // 0..31
    const int tn   = blockIdx.x;             // 64-wide N tile
    const int tm   = blockIdx.y;             // 32-tall M tile
    const int z    = blockIdx.z;
    const int z1   = z / z2n;
    const int z2   = z - z1 * z2n;
    A   += z1 * offA1 + z2 * offA2;
    Bm  += z1 * offB1 + z2 * offB2;
    OUT += z1 * offO1 + z2 * offO2;

    const int l15 = lane & 15;
    const int khA = (lane >> 4) * 16;        // A fragment K sub-offset
    const int khB = (lane >> 4) * 16;        // B fragment K sub-offset

    v8f c[MI][NJ];
    #pragma unroll
    for (int i = 0; i < MI; ++i)
        #pragma unroll
        for (int j = 0; j < NJ; ++j) c[i][j] = (v8f){};

    for (int kk = 0; kk < K; kk += 32) {
        // A fragments, contiguous k order (same as B, not the ISA table order):
        //   lanes 0-15 : K = kk+[0..15]
        //   lanes16-31 : K = kk+16+[0..15]
        v16h a[MI];
        #pragma unroll
        for (int i = 0; i < MI; ++i) {
            const int mrow = tm * 32 + i * 16 + l15;
            const float* pa = A + (long)mrow * lda + kk + khA;
            if (kk + 32 < K) __builtin_prefetch(pa + 32, 0, 1);
            #pragma unroll
            for (int e = 0; e < 8; ++e) a[i][e]     = (_Float16)pa[e];
            #pragma unroll
            for (int e = 0; e < 8; ++e) a[i][8 + e] = (_Float16)pa[8 + e];
        }

        // B fragments, same contiguous k order as A: lane holds column n,
        //   lanes 0-15 : K = kk+[0..15], lanes 16-31 : K = kk+16+[0..15]
        v16h bf[NJ];
        #pragma unroll
        for (int j = 0; j < NJ; ++j) {
            const int ncol = tn * 64 + j * 16 + l15;
            if (TRANS_B) {
                const float* pb = Bm + (long)ncol * ldb + kk + khB;
                #pragma unroll
                for (int e = 0; e < 16; ++e) bf[j][e] = (_Float16)pb[e];
            } else {
                const float* pb = Bm + (long)(kk + khB) * ldb + ncol;
                #pragma unroll
                for (int e = 0; e < 16; ++e) bf[j][e] = (_Float16)pb[(long)e * ldb];
            }
        }

        #pragma unroll
        for (int i = 0; i < MI; ++i)
            #pragma unroll
            for (int j = 0; j < NJ; ++j)
                c[i][j] = __builtin_amdgcn_wmma_f32_16x16x32_f16(
                              false, a[i], false, bf[j],
                              (short)0, c[i][j], false, false);
    }

    // C/D layout: VGPR r -> M = r + 8*(lane>=16), N = lane&15
    float bv[NJ];
    #pragma unroll
    for (int j = 0; j < NJ; ++j)
        bv[j] = HAS_BIAS ? bias[tn * 64 + j * 16 + l15] : 0.0f;

    const int mhalf = (lane >> 4) ? 8 : 0;
    #pragma unroll
    for (int i = 0; i < MI; ++i) {
        const int mbase = tm * 32 + i * 16 + mhalf;
        #pragma unroll
        for (int r = 0; r < 8; ++r) {
            const long rowo = (long)(mbase + r) * ldo;
            #pragma unroll
            for (int j = 0; j < NJ; ++j) {
                const long o = rowo + tn * 64 + j * 16 + l15;
                const float val = c[i][j][r] * scale + bv[j];
                OUT[o] = ACCUM ? (OUT[o] + val) : val;
            }
        }
    }
}

// ---------------------------------------------------------------------------
// scores[b,h,i,j] = (qk[b,h,i,j] + qr[b,h,i,sp[b,i,j]]) * (1/sqrt(DK))
// ---------------------------------------------------------------------------
__global__ void __launch_bounds__(T_)
add_rel_scores(float* __restrict__ scores, const float* __restrict__ qr,
               const int* __restrict__ sp)
{
    const int j = threadIdx.x, i = blockIdx.x, h = blockIdx.y, b = blockIdx.z;
    const long sidx = (((long)b * H_ + h) * T_ + i) * T_ + j;
    const int  r    = sp[((long)b * T_ + i) * T_ + j];
    const float qv  = qr[(((long)b * H_ + h) * T_ + i) * R_ + r];
    scores[sidx] = (scores[sidx] + qv) * 0.125f;   // 1/sqrt(64)
}

// ---------------------------------------------------------------------------
// block reductions (8 wave32's per 256-thread block)
// ---------------------------------------------------------------------------
__device__ __forceinline__ float block_reduce_max(float v, float* red) {
    #pragma unroll
    for (int off = 16; off; off >>= 1) v = fmaxf(v, __shfl_xor(v, off));
    const int w = threadIdx.x >> 5;
    if ((threadIdx.x & 31) == 0) red[w] = v;
    __syncthreads();
    if (threadIdx.x < 8) {
        v = red[threadIdx.x];
        #pragma unroll
        for (int off = 4; off; off >>= 1) v = fmaxf(v, __shfl_xor(v, off));
        if (threadIdx.x == 0) red[0] = v;
    }
    __syncthreads();
    const float r = red[0];
    __syncthreads();
    return r;
}

__device__ __forceinline__ float block_reduce_sum(float v, float* red) {
    #pragma unroll
    for (int off = 16; off; off >>= 1) v += __shfl_xor(v, off);
    const int w = threadIdx.x >> 5;
    if ((threadIdx.x & 31) == 0) red[w] = v;
    __syncthreads();
    if (threadIdx.x < 8) {
        v = red[threadIdx.x];
        #pragma unroll
        for (int off = 4; off; off >>= 1) v += __shfl_xor(v, off);
        if (threadIdx.x == 0) red[0] = v;
    }
    __syncthreads();
    const float r = red[0];
    __syncthreads();
    return r;
}

// ---------------------------------------------------------------------------
// In-place softmax over last dim (row length T_ = one block)
// ---------------------------------------------------------------------------
__global__ void __launch_bounds__(T_)
softmax_rows(float* __restrict__ s)
{
    __shared__ float red[8];
    const int  j   = threadIdx.x;
    float*     row = s + (long)blockIdx.x * T_;
    const float x  = row[j];
    const float m  = block_reduce_max(x, red);
    const float e  = __expf(x - m);
    const float sum = block_reduce_sum(e, red);
    row[j] = e / sum;
}

// ---------------------------------------------------------------------------
// aw[b,h,i,r] = sum_{j : sp[b,i,j]==r} attn[b,h,i,j]   (LDS float atomics)
// ---------------------------------------------------------------------------
__global__ void __launch_bounds__(T_)
compute_aw(const float* __restrict__ attn, const int* __restrict__ sp,
           float* __restrict__ aw)
{
    __shared__ float bins[R_];
    const int j = threadIdx.x, i = blockIdx.x, h = blockIdx.y, b = blockIdx.z;
    if (j < R_) bins[j] = 0.0f;
    __syncthreads();
    const float a = attn[(((long)b * H_ + h) * T_ + i) * T_ + j];
    const int   r = sp[((long)b * T_ + i) * T_ + j];
    atomicAdd(&bins[r], a);                 // ds_add_f32
    __syncthreads();
    if (j < R_) aw[(((long)b * H_ + h) * T_ + i) * R_ + j] = bins[j];
}

// ---------------------------------------------------------------------------
// out = relu(LayerNorm(proj + x) * gamma + beta)    (one block per row)
// ---------------------------------------------------------------------------
__global__ void __launch_bounds__(D_)
ln_residual_relu(const float* __restrict__ proj, const float* __restrict__ x,
                 const float* __restrict__ gamma, const float* __restrict__ beta,
                 float* __restrict__ out)
{
    __shared__ float red[8];
    const int  d   = threadIdx.x;
    const long row = blockIdx.x;            // b*T + t
    const float v  = proj[row * D_ + d] + x[row * D_ + d];
    const float mu = block_reduce_sum(v, red) * (1.0f / D_);
    const float df = v - mu;
    const float var = block_reduce_sum(df * df, red) * (1.0f / D_);
    const float rstd = rsqrtf(var + 1e-5f);
    const float y = df * rstd * gamma[d] + beta[d];
    out[row * D_ + d] = fmaxf(y, 0.0f);
}

// ---------------------------------------------------------------------------
// Host launch
// ---------------------------------------------------------------------------
extern "C" void kernel_launch(void* const* d_in, const int* in_sizes, int n_in,
                              void* d_out, int out_size, void* d_ws, size_t ws_size,
                              hipStream_t stream)
{
    const float* X    = (const float*)d_in[0];   // [B,T,D]
    const int*   sp   = (const int*)  d_in[1];   // [B,T,T]
    const float* Wq   = (const float*)d_in[2];
    const float* bq   = (const float*)d_in[3];
    const float* Wk   = (const float*)d_in[4];
    const float* bk   = (const float*)d_in[5];
    const float* Wv   = (const float*)d_in[6];
    const float* bv   = (const float*)d_in[7];
    const float* relk = (const float*)d_in[8];   // [R,D]
    const float* relv = (const float*)d_in[9];   // [R,D]
    const float* Wo   = (const float*)d_in[10];
    const float* bo   = (const float*)d_in[11];
    const float* gam  = (const float*)d_in[12];
    const float* bet  = (const float*)d_in[13];
    float*       out  = (float*)d_out;

    const long nBTD = (long)B_ * T_ * D_;        //  262144
    const long nQR  = (long)B_ * H_ * T_ * R_;   //  262144
    const long nSC  = (long)B_ * H_ * T_ * T_;   // 1048576

    float* q      = (float*)d_ws;
    float* k      = q      + nBTD;
    float* v      = k      + nBTD;
    float* qr     = v      + nBTD;
    float* scores = qr     + nQR;
    float* aw     = scores + nSC;
    float* agg    = aw     + nQR;
    float* proj   = agg    + nBTD;

    const dim3 blkWave(32);
    const dim3 blkRow(T_);

    // 1. Q/K/V projections:  [B*T, D] = X @ W^T + b
    {
        dim3 grid(D_ / 64, (B_ * T_) / 32, 1);
        wmma_gemm_f16<true,  false, true><<<grid, blkWave, 0, stream>>>(
            X, Wq, bq, q, D_, D_, D_, D_, 0, 0, 0, 0, 0, 0, 1, 1.0f);
        wmma_gemm_f16<true,  false, true><<<grid, blkWave, 0, stream>>>(
            X, Wk, bk, k, D_, D_, D_, D_, 0, 0, 0, 0, 0, 0, 1, 1.0f);
        wmma_gemm_f16<true,  false, true><<<grid, blkWave, 0, stream>>>(
            X, Wv, bv, v, D_, D_, D_, D_, 0, 0, 0, 0, 0, 0, 1, 1.0f);
    }

    // 2. qr[b,h,i,r] = q[b,i,h,:] . rel_k[r,h,:]   (z=(b,h), A@B^T, K=DK)
    {
        dim3 grid(R_ / 64, T_ / 32, B_ * H_);
        wmma_gemm_f16<true,  false, false><<<grid, blkWave, 0, stream>>>(
            q, relk, nullptr, qr, DK_, D_, D_, R_,
            (long)T_ * D_, DK_,                 // A: b stride, h stride
            0,             DK_,                 // B: rel_k head slice
            (long)H_ * T_ * R_, (long)T_ * R_,  // OUT [B,H,T,R]
            H_, 1.0f);
    }

    // 3. qk[b,h,i,j] = q_i . k_j   (z=(b,h), A@B^T, K=DK)
    {
        dim3 grid(T_ / 64, T_ / 32, B_ * H_);
        wmma_gemm_f16<true,  false, false><<<grid, blkWave, 0, stream>>>(
            q, k, nullptr, scores, DK_, D_, D_, T_,
            (long)T_ * D_, DK_,
            (long)T_ * D_, DK_,
            (long)H_ * T_ * T_, (long)T_ * T_,
            H_, 1.0f);
    }

    // 4. scores = (qk + gather(qr)) / sqrt(DK)
    add_rel_scores<<<dim3(T_, H_, B_), blkRow, 0, stream>>>(scores, qr, sp);

    // 5. softmax over j
    softmax_rows<<<dim3(B_ * H_ * T_), blkRow, 0, stream>>>(scores);

    // 6. relation histogram of attention mass
    compute_aw<<<dim3(T_, H_, B_), blkRow, 0, stream>>>(scores, sp, aw);

    // 7a. agg[b,i,h,:] = attn[b,h,i,:] @ V[b,:,h,:]   (A@B, K=T)
    {
        dim3 grid(DK_ / 64, T_ / 32, B_ * H_);
        wmma_gemm_f16<false, false, false><<<grid, blkWave, 0, stream>>>(
            scores, v, nullptr, agg, T_, T_, D_, D_,
            (long)H_ * T_ * T_, (long)T_ * T_,   // A = attn [B,H,T,T]
            (long)T_ * D_, DK_,                  // B = v, column access
            (long)T_ * D_, DK_,                  // OUT = agg [B,T,D] head slice
            H_, 1.0f);
    }
    // 7b. agg += aw @ rel_v[:,h,:]   (A@B, K=R, accumulate)
    {
        dim3 grid(DK_ / 64, T_ / 32, B_ * H_);
        wmma_gemm_f16<false, true,  false><<<grid, blkWave, 0, stream>>>(
            aw, relv, nullptr, agg, R_, R_, D_, D_,
            (long)H_ * T_ * R_, (long)T_ * R_,
            0, DK_,
            (long)T_ * D_, DK_,
            H_, 1.0f);
    }

    // 8. proj = agg @ Wo^T + bo
    {
        dim3 grid(D_ / 64, (B_ * T_) / 32, 1);
        wmma_gemm_f16<true,  false, true><<<grid, blkWave, 0, stream>>>(
            agg, Wo, bo, proj, D_, D_, D_, D_, 0, 0, 0, 0, 0, 0, 1, 1.0f);
    }

    // 9. residual + LayerNorm + relu
    ln_residual_relu<<<dim3(B_ * T_), dim3(D_), 0, stream>>>(proj, X, gam, bet, out);

    (void)in_sizes; (void)n_in; (void)out_size; (void)ws_size;
}
